// TransformerProjector_53171695125249
// MI455X (gfx1250) — matrix operand order repaired and checked by reading, not compiled, not run on hardware
//
#include <hip/hip_runtime.h>
#include <math.h>

// ---------------------------------------------------------------------------
// MI455X (gfx1250, wave32) transformer forward.
// All large GEMMs run on v_wmma_f32_16x16x32_bf16 with fp32->bf16 conversion
// fused into the LDS staging path. Attention (banded, <=121 keys) uses a
// register-only online-softmax, one wave per (batch, query, head).
// ---------------------------------------------------------------------------

typedef __attribute__((ext_vector_type(16))) __bf16 v16bf;
typedef __attribute__((ext_vector_type(8)))  float  v8f;
typedef __attribute__((ext_vector_type(4)))  float  f4;

#define SEQ     1024      // S = T / STRIDE
#define MROWS   4096      // B * S
#define DMODEL  1024
#define DFF     4096
#define NHEADS  8
#define DHEAD   128
#define NLAYERS 4

#define EPI_NONE 0
#define EPI_SILU 1

// ---------------------------------------------------------------------------
// Tiled bf16 WMMA GEMM:  C[M,N] = A[M,K] @ W[K,N] + bias (+SiLU) (+residual)
// Block = 256 threads = 8 waves, tile 128x128, K step 32.
// Waves arranged 4(M) x 2(N): each wave owns 32x64 = 2x4 WMMA tiles.
// ---------------------------------------------------------------------------
__global__ __launch_bounds__(256)
void gemm_bf16_wmma(const float* __restrict__ A, const float* __restrict__ W,
                    const float* __restrict__ bias,
                    const float* __restrict__ residual,
                    float* __restrict__ C,
                    int M, int N, int K, int epi)
{
    __shared__ alignas(16) __bf16 As[128 * 32];   // [row][k]   8 KB
    __shared__ alignas(16) __bf16 Bs[128 * 32];   // [n][k] transposed, 8 KB

    const int tid  = threadIdx.x;
    const int lane = tid & 31;
    const int wid  = tid >> 5;        // 0..7
    const int wm   = wid & 3;         // 0..3 -> 32-row slab
    const int wn   = wid >> 2;        // 0..1 -> 64-col slab

    const int blockN = blockIdx.x * 128;
    const int blockM = blockIdx.y * 128;

    v8f acc[2][4];
    #pragma unroll
    for (int i = 0; i < 2; ++i)
        #pragma unroll
        for (int j = 0; j < 4; ++j)
            #pragma unroll
            for (int r = 0; r < 8; ++r)
                acc[i][j][r] = 0.0f;

    // Staging thread mapping (constant across K loop)
    const int arow  = tid >> 1;             // 0..127
    const int aseg  = (tid & 1) * 16;       // 0 or 16 (floats)
    const int wkk   = tid >> 3;             // 0..31
    const int wnb   = (tid & 7) * 16;       // 0..112

    const float* aptr = A + (size_t)(blockM + arow) * K + aseg;
    const float* wptr = W + (size_t)wkk * N + blockN + wnb;

    for (int k0 = 0; k0 < K; k0 += 32) {
        // ---- stage A tile (128x32 fp32 -> bf16), 16 floats per thread ----
        {
            const f4* src = (const f4*)(aptr + k0);
            f4 f0 = src[0], f1 = src[1], f2 = src[2], f3 = src[3];
            union { __bf16 h[16]; f4 f[2]; } t;
            t.h[0]=(__bf16)f0.x; t.h[1]=(__bf16)f0.y; t.h[2]=(__bf16)f0.z; t.h[3]=(__bf16)f0.w;
            t.h[4]=(__bf16)f1.x; t.h[5]=(__bf16)f1.y; t.h[6]=(__bf16)f1.z; t.h[7]=(__bf16)f1.w;
            t.h[8]=(__bf16)f2.x; t.h[9]=(__bf16)f2.y; t.h[10]=(__bf16)f2.z; t.h[11]=(__bf16)f2.w;
            t.h[12]=(__bf16)f3.x; t.h[13]=(__bf16)f3.y; t.h[14]=(__bf16)f3.z; t.h[15]=(__bf16)f3.w;
            f4* dst = (f4*)&As[arow * 32 + aseg];
            dst[0] = t.f[0];
            dst[1] = t.f[1];
            if (k0 + 32 < K)
                __builtin_prefetch((const void*)(aptr + k0 + 32), 0, 0);
        }
        // ---- stage W tile (32x128 fp32 -> bf16 transposed [n][k]) ----
        {
            const f4* src = (const f4*)(wptr + (size_t)k0 * N);
            f4 w0 = src[0], w1 = src[1], w2 = src[2], w3 = src[3];
            float wv[16] = { w0.x,w0.y,w0.z,w0.w, w1.x,w1.y,w1.z,w1.w,
                             w2.x,w2.y,w2.z,w2.w, w3.x,w3.y,w3.z,w3.w };
            #pragma unroll
            for (int i = 0; i < 16; ++i)
                Bs[(wnb + i) * 32 + wkk] = (__bf16)wv[i];
            if (k0 + 32 < K)
                __builtin_prefetch((const void*)(wptr + (size_t)(k0 + 32) * N), 0, 0);
        }
        __syncthreads();

        // ---- fragments per CDNA5 16-bit A/B VGPR layouts ----
        const int aklo  = (lane >> 4) * 8;     // A: lanes 16-31 hold K+8
        const int bklo  = (lane >> 4) * 8;     // B: lanes 16-31 hold K+8
        const int arow0 = wm * 32 + (lane & 15);
        const int bcol0 = wn * 64 + (lane & 15);

        v16bf afrag[2], bfrag[4];
        #pragma unroll
        for (int tm = 0; tm < 2; ++tm) {
            const int r = arow0 + tm * 16;
            union { v16bf v; f4 f[2]; } u;
            u.f[0] = *(const f4*)&As[r * 32 + aklo];
            u.f[1] = *(const f4*)&As[r * 32 + 16 + aklo];
            afrag[tm] = u.v;
        }
        #pragma unroll
        for (int tn = 0; tn < 4; ++tn) {
            const int c = bcol0 + tn * 16;
            union { v16bf v; f4 f[2]; } u;
            u.f[0] = *(const f4*)&Bs[c * 32 + bklo];
            u.f[1] = *(const f4*)&Bs[c * 32 + 16 + bklo];
            bfrag[tn] = u.v;
        }
        #pragma unroll
        for (int tm = 0; tm < 2; ++tm)
            #pragma unroll
            for (int tn = 0; tn < 4; ++tn)
                acc[tm][tn] = __builtin_amdgcn_wmma_f32_16x16x32_bf16(
                    false, afrag[tm], false, bfrag[tn],
                    (short)0, acc[tm][tn], false, false);
        __syncthreads();
    }

    // ---- epilogue: bias (+SiLU) (+residual), fp32 store ----
    // C/D layout: VGPR r -> M = r (lanes 0-15) / r+8 (lanes 16-31); N = lane%16
    #pragma unroll
    for (int tm = 0; tm < 2; ++tm) {
        #pragma unroll
        for (int tn = 0; tn < 4; ++tn) {
            const int gmBase = blockM + wm * 32 + tm * 16 + ((lane >> 4) * 8);
            const int gn     = blockN + wn * 64 + tn * 16 + (lane & 15);
            const float bval = bias[gn];
            #pragma unroll
            for (int r = 0; r < 8; ++r) {
                const int gm = gmBase + r;
                float val = acc[tm][tn][r] + bval;
                if (epi == EPI_SILU)
                    val = val / (1.0f + __expf(-val));
                if (residual)
                    val += residual[(size_t)gm * N + gn];
                C[(size_t)gm * N + gn] = val;
            }
        }
    }
}

// ---------------------------------------------------------------------------
// LayerNorm over 1024-wide rows; one 256-thread block per row, fp32 out.
// ---------------------------------------------------------------------------
__global__ __launch_bounds__(256)
void layernorm_kernel(const float* __restrict__ x, const float* __restrict__ s,
                      const float* __restrict__ b, float* __restrict__ out)
{
    __shared__ float red[8];
    const int row = blockIdx.x;
    const int tid = threadIdx.x;
    const float* xr = x + (size_t)row * DMODEL;

    f4 v = ((const f4*)xr)[tid];
    float sum = v.x + v.y + v.z + v.w;
    #pragma unroll
    for (int o = 16; o >= 1; o >>= 1) sum += __shfl_xor(sum, o, 32);
    if ((tid & 31) == 0) red[tid >> 5] = sum;
    __syncthreads();
    float tot = 0.0f;
    #pragma unroll
    for (int i = 0; i < 8; ++i) tot += red[i];
    const float mu = tot * (1.0f / (float)DMODEL);
    __syncthreads();

    const float dx = v.x - mu, dy = v.y - mu, dz = v.z - mu, dw = v.w - mu;
    float sq = dx*dx + dy*dy + dz*dz + dw*dw;
    #pragma unroll
    for (int o = 16; o >= 1; o >>= 1) sq += __shfl_xor(sq, o, 32);
    if ((tid & 31) == 0) red[tid >> 5] = sq;
    __syncthreads();
    float var = 0.0f;
    #pragma unroll
    for (int i = 0; i < 8; ++i) var += red[i];
    var *= (1.0f / (float)DMODEL);
    const float inv = rsqrtf(var + 1e-5f);

    const int c = tid * 4;
    float* outr = out + (size_t)row * DMODEL;
    outr[c + 0] = dx * inv * s[c + 0] + b[c + 0];
    outr[c + 1] = dy * inv * s[c + 1] + b[c + 1];
    outr[c + 2] = dz * inv * s[c + 2] + b[c + 2];
    outr[c + 3] = dw * inv * s[c + 3] + b[c + 3];
}

// ---------------------------------------------------------------------------
// RoPE in-place on q and k: one thread per (row, head, freq). 2M threads.
// ---------------------------------------------------------------------------
__global__ __launch_bounds__(256)
void rope_kernel(float* __restrict__ q, float* __restrict__ k)
{
    const int tid = blockIdx.x * blockDim.x + threadIdx.x;  // [0, 4096*512)
    const int f   = tid & 63;
    const int hh  = (tid >> 6) & 7;
    const int row = tid >> 9;
    const int pos = row & (SEQ - 1);

    // inv_freq = 10000^(-f/64) = exp(-f * ln(10000)/64)
    const float inv_freq = __expf(-(float)f * 0.14391156831213413f);
    const float ang = (float)pos * inv_freq;
    const float c = cosf(ang), sn = sinf(ang);

    const size_t base = (size_t)row * DMODEL + hh * DHEAD + f;
    float x1 = q[base], x2 = q[base + 64];
    q[base]      = x1 * c - x2 * sn;
    q[base + 64] = x1 * sn + x2 * c;
    x1 = k[base]; x2 = k[base + 64];
    k[base]      = x1 * c - x2 * sn;
    k[base + 64] = x1 * sn + x2 * c;
}

// ---------------------------------------------------------------------------
// Zero rows beyond sequence length (pad mask after input projection).
// ---------------------------------------------------------------------------
__global__ __launch_bounds__(256)
void padmask_kernel(float* __restrict__ h, const int* __restrict__ lens)
{
    const int tid = blockIdx.x * blockDim.x + threadIdx.x;   // [0, 4M)
    const int row = tid >> 10;
    const int s   = row & (SEQ - 1);
    const int b   = row >> 10;
    if (s >= lens[b] / 4) h[tid] = 0.0f;
}

// ---------------------------------------------------------------------------
// Banded attention, one wave per (b, qi, h): online softmax, register-only.
// Band: j in [qi-99, qi+20]. Keys with j >= valid get -1e9 bias (matching
// the reference, which softmaxes the -1e9 bias -> uniform if fully masked).
// ctx may alias q (each wave reads only its own q slice before writing).
// ---------------------------------------------------------------------------
__global__ __launch_bounds__(256)
void attn_kernel(const float* __restrict__ q, const float* __restrict__ k,
                 const float* __restrict__ v, const int* __restrict__ lens,
                 float* __restrict__ ctx)
{
    const int wglobal = blockIdx.x * 8 + (threadIdx.x >> 5);
    const int lane = threadIdx.x & 31;
    const int hh   = wglobal & 7;
    const int row  = wglobal >> 3;        // b*SEQ + qi
    const int qi   = row & (SEQ - 1);
    const int b    = row >> 10;
    const int valid = lens[b] / 4;

    const size_t qoff = (size_t)row * DMODEL + hh * DHEAD + lane * 4;
    const f4 qv = *(const f4*)&q[qoff];
    const float scale = 0.08838834764831845f;   // 1/sqrt(128)

    int jlo = qi - 99; if (jlo < 0) jlo = 0;
    int jhi = qi + 20; if (jhi > SEQ - 1) jhi = SEQ - 1;

    float m = -3.0e38f, ssum = 0.0f;
    f4 acc; acc.x = acc.y = acc.z = acc.w = 0.0f;

    for (int j = jlo; j <= jhi; ++j) {
        const size_t koff = (size_t)((b << 10) | j) * DMODEL + hh * DHEAD + lane * 4;
        const f4 kv = *(const f4*)&k[koff];
        float d = qv.x * kv.x + qv.y * kv.y + qv.z * kv.z + qv.w * kv.w;
        d += __shfl_xor(d, 16, 32);
        d += __shfl_xor(d,  8, 32);
        d += __shfl_xor(d,  4, 32);
        d += __shfl_xor(d,  2, 32);
        d += __shfl_xor(d,  1, 32);
        d = d * scale + (j < valid ? 0.0f : -1.0e9f);

        const float mn   = fmaxf(m, d);
        const float corr = __expf(m - mn);
        const float p    = __expf(d - mn);
        ssum = ssum * corr + p;
        const f4 vv = *(const f4*)&v[koff];
        acc.x = acc.x * corr + p * vv.x;
        acc.y = acc.y * corr + p * vv.y;
        acc.z = acc.z * corr + p * vv.z;
        acc.w = acc.w * corr + p * vv.w;
        m = mn;
    }
    const float invs = 1.0f / ssum;
    f4 o; o.x = acc.x * invs; o.y = acc.y * invs; o.z = acc.z * invs; o.w = acc.w * invs;
    *(f4*)&ctx[qoff] = o;
}

// ---------------------------------------------------------------------------
// Host orchestration
// ---------------------------------------------------------------------------
extern "C" void kernel_launch(void* const* d_in, const int* in_sizes, int n_in,
                              void* d_out, int out_size, void* d_ws, size_t ws_size,
                              hipStream_t stream)
{
    (void)in_sizes; (void)n_in; (void)out_size; (void)ws_size;

    const float* x    = (const float*)d_in[0];
    const int*   xlen = (const int*)  d_in[1];
    const float* Wp   = (const float*)d_in[2];
    const float* bp   = (const float*)d_in[3];
    const float* ln1s = (const float*)d_in[4];
    const float* ln1b = (const float*)d_in[5];
    const float* Wq   = (const float*)d_in[6];
    const float* bq   = (const float*)d_in[7];
    const float* Wk   = (const float*)d_in[8];
    const float* bk   = (const float*)d_in[9];
    const float* Wv   = (const float*)d_in[10];
    const float* bv   = (const float*)d_in[11];
    const float* Wo   = (const float*)d_in[12];
    const float* bo   = (const float*)d_in[13];
    const float* ln2s = (const float*)d_in[14];
    const float* ln2b = (const float*)d_in[15];
    const float* W1   = (const float*)d_in[16];
    const float* b1   = (const float*)d_in[17];
    const float* W2   = (const float*)d_in[18];
    const float* b2   = (const float*)d_in[19];
    const float* lnfs = (const float*)d_in[20];
    const float* lnfb = (const float*)d_in[21];

    // Workspace layout (floats). ffn overlaps the dead k/v buffers.
    //   h[4M] | y[4M] | q[4M] | k[4M] | v[4M]
    //                          \------ ffn[16M] ------/   total 28M fl = 112MB
    const size_t MD = (size_t)MROWS * DMODEL;   // 4M floats
    float* h   = (float*)d_ws;
    float* y   = h  + MD;
    float* qb  = y  + MD;
    float* kb  = qb + MD;
    float* vb  = kb + MD;
    float* ffn = kb;                            // 16M floats (overlaps k,v + tail)

    const dim3 blk(256);
    const dim3 grdD(DMODEL / 128, MROWS / 128);  // N=1024 GEMMs: 8 x 32
    const dim3 grdF(DFF    / 128, MROWS / 128);  // N=4096 GEMM : 32 x 32

    // Input projection: h = xf @ Wp + bp   (xf is x reinterpreted as 4096x2048)
    gemm_bf16_wmma<<<grdD, blk, 0, stream>>>(x, Wp, bp, nullptr, h,
                                             MROWS, DMODEL, 2048, EPI_NONE);
    padmask_kernel<<<(MROWS * DMODEL) / 256, blk, 0, stream>>>(h, xlen);

    for (int l = 0; l < NLAYERS; ++l) {
        const size_t wOff  = (size_t)l * DMODEL * DMODEL;
        const size_t w1Off = (size_t)l * DMODEL * DFF;
        const size_t w2Off = (size_t)l * DFF * DMODEL;

        // y = LN1(h)
        layernorm_kernel<<<MROWS, blk, 0, stream>>>(h, ln1s + l * DMODEL,
                                                    ln1b + l * DMODEL, y);
        // q,k,v projections
        gemm_bf16_wmma<<<grdD, blk, 0, stream>>>(y, Wq + wOff, bq + l * DMODEL,
                                                 nullptr, qb, MROWS, DMODEL, DMODEL, EPI_NONE);
        gemm_bf16_wmma<<<grdD, blk, 0, stream>>>(y, Wk + wOff, bk + l * DMODEL,
                                                 nullptr, kb, MROWS, DMODEL, DMODEL, EPI_NONE);
        gemm_bf16_wmma<<<grdD, blk, 0, stream>>>(y, Wv + wOff, bv + l * DMODEL,
                                                 nullptr, vb, MROWS, DMODEL, DMODEL, EPI_NONE);
        // RoPE on q,k
        rope_kernel<<<(MROWS * NHEADS * 64) / 256, blk, 0, stream>>>(qb, kb);
        // Banded attention; ctx written in place over q
        attn_kernel<<<MROWS * NHEADS / 8, blk, 0, stream>>>(qb, kb, vb, xlen, qb);
        // h = h + ctx @ Wo + bo
        gemm_bf16_wmma<<<grdD, blk, 0, stream>>>(qb, Wo + wOff, bo + l * DMODEL,
                                                 h, h, MROWS, DMODEL, DMODEL, EPI_NONE);
        // y = LN2(h)
        layernorm_kernel<<<MROWS, blk, 0, stream>>>(h, ln2s + l * DMODEL,
                                                    ln2b + l * DMODEL, y);
        // ffn = silu(y @ W1 + b1)
        gemm_bf16_wmma<<<grdF, blk, 0, stream>>>(y, W1 + w1Off, b1 + l * DFF,
                                                 nullptr, ffn, MROWS, DFF, DMODEL, EPI_SILU);
        // h = h + ffn @ W2 + b2
        gemm_bf16_wmma<<<grdD, blk, 0, stream>>>(ffn, W2 + w2Off, b2 + l * DMODEL,
                                                 h, h, MROWS, DMODEL, DFF, EPI_NONE);
    }
    // out = LN_f(h)
    layernorm_kernel<<<MROWS, blk, 0, stream>>>(h, lnfs, lnfb, (float*)d_out);
}
